// LSTM_25890062860556
// MI455X (gfx1250) — matrix operand order repaired and checked by reading, not compiled, not run on hardware
//
#include <hip/hip_runtime.h>
#include <stdint.h>

typedef __attribute__((ext_vector_type(16))) __bf16 v16bf;
typedef __attribute__((ext_vector_type(8)))  float  v8f;
typedef unsigned short u16;

#define NN 20000       // nodes
#define NE 640000      // edges
#define D  128         // feature dim (in == out)
#define COLS 512       // 4 gates * 128
#define ND (NN * D)

__device__ __forceinline__ u16 f2bf(float f) {
    union { float f; unsigned int u; } c; c.f = f;
    unsigned int u = c.u;
    u += 0x7FFFu + ((u >> 16) & 1u);   // round-to-nearest-even
    return (u16)(u >> 16);
}

__global__ void zero_kernel(float* p, int n) {
    int i = blockIdx.x * blockDim.x + threadIdx.x;
    if (i < n) p[i] = 0.f;
}

// One wave (32 lanes) per edge; lane handles 4 consecutive features.
__global__ __launch_bounds__(256)
void edge_scatter_kernel(const float* __restrict__ X, const float* __restrict__ H,
                         const int* __restrict__ ei, const float* __restrict__ ew,
                         float* __restrict__ aggX, float* __restrict__ aggH,
                         float* __restrict__ cnt) {
    int e = blockIdx.x * 8 + (threadIdx.x >> 5);
    if (e >= NE) return;
    int lane = threadIdx.x & 31;
    int src = ei[e];
    int dst = ei[NE + e];
    float w = ew[e];
    float4 xv = *(reinterpret_cast<const float4*>(X + (size_t)src * D) + lane);
    float4 hv = *(reinterpret_cast<const float4*>(H + (size_t)src * D) + lane);
    float* ax = aggX + (size_t)dst * D + lane * 4;
    float* ah = aggH + (size_t)dst * D + lane * 4;
    atomicAdd(ax + 0, xv.x * w); atomicAdd(ax + 1, xv.y * w);
    atomicAdd(ax + 2, xv.z * w); atomicAdd(ax + 3, xv.w * w);
    atomicAdd(ah + 0, hv.x * w); atomicAdd(ah + 1, hv.y * w);
    atomicAdd(ah + 2, hv.z * w); atomicAdd(ah + 3, hv.w * w);
    if (lane == 0) atomicAdd(&cnt[dst], 1.0f);
}

__global__ __launch_bounds__(256)
void normalize_convert_kernel(const float* __restrict__ X, const float* __restrict__ H,
                              const float* __restrict__ aggX, const float* __restrict__ aggH,
                              const float* __restrict__ cnt,
                              u16* __restrict__ Xb, u16* __restrict__ Hb,
                              u16* __restrict__ aggXb, u16* __restrict__ aggHb) {
    int i = blockIdx.x * blockDim.x + threadIdx.x;
    if (i >= ND) return;
    int n = i >> 7;
    float inv = 1.f / fmaxf(cnt[n], 1.f);
    Xb[i]    = f2bf(X[i]);
    Hb[i]    = f2bf(H[i]);
    aggXb[i] = f2bf(aggX[i] * inv);
    aggHb[i] = f2bf(aggH[i] * inv);
}

// Pre-swizzle weights into per-lane WMMA B-fragment order:
// Wfrag[((s*32+ct)*4+kt)*32 + lane][16 bf16], lane: n=lane&15, hs=lane>>4,
// element j -> (k = kt*32 + hs*16 + j, col = ct*16 + n), col = g*128 + o.
__global__ __launch_bounds__(256)
void pack_weights_kernel(const float* __restrict__ Wxl, const float* __restrict__ Wxr,
                         const float* __restrict__ Whl, const float* __restrict__ Whr,
                         u16* __restrict__ Wfrag) {
    int tid = blockIdx.x * blockDim.x + threadIdx.x;
    if (tid >= 4 * 32 * 4 * 32) return;
    int lane = tid & 31;
    int kt   = (tid >> 5) & 3;
    int ct   = (tid >> 7) & 31;
    int s    = tid >> 12;
    const float* W = (s == 0) ? Wxl : (s == 1) ? Wxr : (s == 2) ? Whl : Whr;
    int n  = lane & 15;
    int hs = lane >> 4;
    int col = ct * 16 + n;       // 0..511
    int g = col >> 7, o = col & 127;
    u16* dst = Wfrag + (size_t)tid * 16;
    for (int j = 0; j < 16; ++j) {
        int k = kt * 32 + hs * 16 + j;               // input dim index d
        dst[j] = f2bf(W[((size_t)g * D + k) * D + o]); // W[g][d][o]
    }
}

__global__ void bias_kernel(const float* __restrict__ bx, const float* __restrict__ bh,
                            const float* __restrict__ bg, float* __restrict__ bsum) {
    int i = blockIdx.x * blockDim.x + threadIdx.x;
    if (i < COLS) bsum[i] = bx[i] + bh[i] + bg[i];
}

__device__ __forceinline__ float sigm(float x) { return 1.f / (1.f + __expf(-x)); }

// One block per 16-row tile. 4 waves; each wave handles 8 of the 32 column
// tiles (512 cols = 4 gates * 128). 16 v_wmma_f32_16x16x32_bf16 per tile.
__global__ __launch_bounds__(128)
void gclstm_gemm_kernel(const u16* __restrict__ aggXb, const u16* __restrict__ Xb,
                        const u16* __restrict__ aggHb, const u16* __restrict__ Hb,
                        const u16* __restrict__ Wfrag, const float* __restrict__ bsum,
                        const float* __restrict__ Cin, const float* __restrict__ wc,
                        float* __restrict__ Hout, float* __restrict__ Cout) {
    __shared__ float Gt[16][COLS];   // 32 KB
    const int row0 = blockIdx.x * 16;
    const int lane = threadIdx.x & 31;
    const int wave = threadIdx.x >> 5;
    const int m  = lane & 15;
    const int hs = lane >> 4;

    const u16* Asrc[4] = { aggXb, Xb, aggHb, Hb };

    for (int q = 0; q < 8; ++q) {
        const int ct = wave * 8 + q;
        v8f acc = {};
#pragma unroll
        for (int s = 0; s < 4; ++s) {
            const u16* arow  = Asrc[s] + (size_t)(row0 + m) * D;
            const u16* wbase = Wfrag + ((size_t)(s * 32 + ct) * 4) * 32 * 16;
#pragma unroll
            for (int kt = 0; kt < 4; ++kt) {
                const int k0 = kt * 32;
                union { v16bf v; uint4 u[2]; } af, bfr;
                // A: contiguous k order, same as Wfrag. lanes 0-15: K = {k0..k0+15}
                //    lanes 16-31: K = {k0+16..k0+31}
                const u16* pa = arow + k0 + hs * 16;
                af.u[0] = *reinterpret_cast<const uint4*>(pa);
                af.u[1] = *reinterpret_cast<const uint4*>(pa + 8);
                // B: pre-swizzled contiguous 32B per lane
                const u16* pb = wbase + ((size_t)kt * 32 + lane) * 16;
                bfr.u[0] = *reinterpret_cast<const uint4*>(pb);
                bfr.u[1] = *reinterpret_cast<const uint4*>(pb + 8);
                acc = __builtin_amdgcn_wmma_f32_16x16x32_bf16(
                          false, af.v, false, bfr.v, (short)0, acc, false, false);
            }
        }
        // C/D layout: lane n = lane&15, VGPR r -> row r + 8*hs
        const int n = lane & 15;
#pragma unroll
        for (int r = 0; r < 8; ++r) {
            int mm = r + 8 * hs;
            int cc = ct * 16 + n;
            Gt[mm][cc] = acc[r] + bsum[cc];
        }
    }
    __syncthreads();

    // Elementwise peephole LSTM over the 16x128 tile (2048 elems / 128 thr)
    for (int t = 0; t < 16; ++t) {
        int idx = threadIdx.x + t * 128;
        int r = idx >> 7;
        int o = idx & 127;
        int node = row0 + r;                 // NN = 1250*16, always in range
        size_t p = (size_t)node * D + o;
        float c_old = Cin[p];
        float gI = Gt[r][0 * D + o];
        float gF = Gt[r][1 * D + o];
        float gC = Gt[r][2 * D + o];
        float gO = Gt[r][3 * D + o];
        float I = sigm(gI + wc[0 * D + o] * c_old);
        float F = sigm(gF + wc[1 * D + o] * c_old);
        float T = tanhf(gC);
        float c_new = F * c_old + I * T;
        float O = sigm(gO + wc[2 * D + o] * c_new);
        Hout[p] = O * tanhf(c_new);
        Cout[p] = c_new;
    }
}

extern "C" void kernel_launch(void* const* d_in, const int* in_sizes, int n_in,
                              void* d_out, int out_size, void* d_ws, size_t ws_size,
                              hipStream_t stream) {
    const float* X   = (const float*)d_in[0];
    const int*   ei  = (const int*)  d_in[1];   // [2, NE]
    const float* ew  = (const float*)d_in[2];
    const float* H   = (const float*)d_in[3];
    const float* C   = (const float*)d_in[4];
    const float* Wxl = (const float*)d_in[5];
    const float* Wxr = (const float*)d_in[6];
    const float* bx  = (const float*)d_in[7];
    const float* Whl = (const float*)d_in[8];
    const float* Whr = (const float*)d_in[9];
    const float* bh  = (const float*)d_in[10];
    const float* wc  = (const float*)d_in[11];
    const float* bg  = (const float*)d_in[12];

    // workspace layout (256B aligned slices)
    char* w = (char*)d_ws;
    size_t off = 0;
    auto take = [&](size_t bytes) { void* p = w + off; off += (bytes + 255) & ~(size_t)255; return p; };
    float* aggX  = (float*)take((size_t)ND * 4);   // contiguous with aggH, cnt
    float* aggH  = (float*)take((size_t)ND * 4);
    float* cnt   = (float*)take((size_t)NN * 4);
    u16*   Xb    = (u16*)  take((size_t)ND * 2);
    u16*   Hb    = (u16*)  take((size_t)ND * 2);
    u16*   aggXb = (u16*)  take((size_t)ND * 2);
    u16*   aggHb = (u16*)  take((size_t)ND * 2);
    u16*   Wfrag = (u16*)  take((size_t)4 * 32 * 4 * 32 * 16 * 2);
    float* bsum  = (float*)take((size_t)COLS * 4);

    float* Hout = (float*)d_out;
    float* Cout = Hout + (size_t)ND;

    // 1) zero aggX|aggH|cnt (contiguous)
    {
        int n = 2 * ND + NN;
        zero_kernel<<<(n + 255) / 256, 256, 0, stream>>>(aggX, n);
    }
    // 2) edge scatter (wave per edge)
    edge_scatter_kernel<<<(NE + 7) / 8, 256, 0, stream>>>(X, H, ei, ew, aggX, aggH, cnt);
    // 3) normalize + bf16 convert
    normalize_convert_kernel<<<(ND + 255) / 256, 256, 0, stream>>>(
        X, H, aggX, aggH, cnt, Xb, Hb, aggXb, aggHb);
    // 4) pack weights into WMMA B-fragment layout
    pack_weights_kernel<<<(4 * 32 * 4 * 32 + 255) / 256, 256, 0, stream>>>(
        Wxl, Wxr, Whl, Whr, Wfrag);
    // 5) fold biases
    bias_kernel<<<2, 256, 0, stream>>>(bx, bh, bg, bsum);
    // 6) fused WMMA GEMM + LSTM gates
    gclstm_gemm_kernel<<<NN / 16, 128, 0, stream>>>(
        aggXb, Xb, aggHb, Hb, Wfrag, bsum, C, wc, Hout, Cout);
}
